// GRUBase_82386062672049
// MI455X (gfx1250) — matrix operand order repaired and checked by reading, not compiled, not run on hardware
//
#include <hip/hip_runtime.h>
#include <stdint.h>

// Problem sizes (match the reference)
#define IN_DIM   1024
#define HID_DIM  1024
#define BATCH    64
#define TSTEPS   512
#define MROWS    (BATCH * TSTEPS)   // 32768 rows for input projections

typedef __attribute__((ext_vector_type(16))) __bf16 v16bf;
typedef __attribute__((ext_vector_type(8)))  float  v8f;
typedef __attribute__((ext_vector_type(4)))  unsigned int v4ui;
typedef __attribute__((ext_vector_type(8)))  int v8si;
typedef __attribute__((ext_vector_type(4)))  int v4si;

#if defined(__has_builtin)
#if __has_builtin(__builtin_amdgcn_tensor_load_to_lds) && \
    __has_builtin(__builtin_amdgcn_s_wait_tensorcnt)
#define USE_TDM 1
#endif
#endif

union Frag32B {
  uint4 u[2];
  v16bf v;
};

__device__ __forceinline__ unsigned short f32_to_bf16_raw(float f) {
  union { float f; unsigned int u; } c; c.f = f;
  // round-to-nearest-even
  return (unsigned short)((c.u + 0x7FFFu + ((c.u >> 16) & 1u)) >> 16);
}

// ---------------------------------------------------------------------------
// fp32 -> bf16 conversion (grid-stride) — used for the activation matrix x
// ---------------------------------------------------------------------------
__global__ void cvt_f32_bf16(const float* __restrict__ src,
                             unsigned short* __restrict__ dst, int n) {
  int i = blockIdx.x * blockDim.x + threadIdx.x;
  int stride = gridDim.x * blockDim.x;
  for (; i < n; i += stride) dst[i] = f32_to_bf16_raw(src[i]);
}

// ---------------------------------------------------------------------------
// fp32 [K][N] -> bf16 [N][K] tiled transpose-convert (weights, one-time).
// Pre-transposing makes all downstream LDS staging coalesced b128 traffic.
// block (32,8), grid (N/32, K/32)
// ---------------------------------------------------------------------------
__global__ __launch_bounds__(256) void transp_cvt_f32_bf16(
    const float* __restrict__ src, unsigned short* __restrict__ dst) {
  __shared__ float tile[32][33];
  const int tx = threadIdx.x;
  const int ty = threadIdx.y;
  const int n0 = blockIdx.x * 32;
  const int k0 = blockIdx.y * 32;
  #pragma unroll
  for (int j = 0; j < 32; j += 8)
    tile[ty + j][tx] = src[(size_t)(k0 + ty + j) * HID_DIM + n0 + tx];
  __syncthreads();
  #pragma unroll
  for (int j = 0; j < 32; j += 8)
    dst[(size_t)(n0 + ty + j) * IN_DIM + k0 + tx] =
        f32_to_bf16_raw(tile[tx][ty + j]);
}

// ---------------------------------------------------------------------------
// Phase 1: G = X(bf16)[32768,1024] @ W(bf16)[1024,1024] + bias  (fp32 out)
// Weights arrive PRE-TRANSPOSED: WT[n][k]. grid=(N/128, M/128, 3), block=256.
// ---------------------------------------------------------------------------
#define BM 128
#define BN 128
#define BK 32
#define APAD 40   // bf16 elems per LDS row (mult of 8 -> 16B aligned, bank-skewed)
#define BPAD 40

__global__ __launch_bounds__(256) void gemm_xproj(
    const unsigned short* __restrict__ X,
    const unsigned short* __restrict__ WrT,
    const unsigned short* __restrict__ WzT,
    const unsigned short* __restrict__ WnT,
    const float* __restrict__ br, const float* __restrict__ bz,
    const float* __restrict__ bn,
    float* __restrict__ Gr, float* __restrict__ Gz, float* __restrict__ Gn)
{
  __shared__ unsigned short ldsA[BM * APAD];
  __shared__ unsigned short ldsBT[BN * BPAD];

  const unsigned short* WT; const float* bias; float* C;
  if (blockIdx.z == 0)      { WT = WrT; bias = br; C = Gr; }
  else if (blockIdx.z == 1) { WT = WzT; bias = bz; C = Gz; }
  else                      { WT = WnT; bias = bn; C = Gn; }

  const int n0    = blockIdx.x * BN;
  const int m0    = blockIdx.y * BM;
  const int tid   = threadIdx.x;
  const int lane  = tid & 31;
  const int wave  = tid >> 5;
  const int wm    = wave & 1;
  const int wn    = wave >> 1;
  const int l16   = lane & 15;
  const int lhalf = lane >> 4;

  v8f acc[4][2];
  {
    v8f vz = {};
    #pragma unroll
    for (int i = 0; i < 4; ++i)
      #pragma unroll
      for (int j = 0; j < 2; ++j) acc[i][j] = vz;
  }

  for (int k0 = 0; k0 < IN_DIM; k0 += BK) {
    // stage A tile [128 x 32] row-major (padded) — pure b128 traffic
    #pragma unroll
    for (int it = 0; it < 2; ++it) {
      int slot = it * 256 + tid;         // 512 slots of 8 bf16 (16B)
      int row  = slot >> 2;
      int quad = slot & 3;
      uint4 vv = *(const uint4*)(X + (size_t)(m0 + row) * IN_DIM + k0 + quad * 8);
      *(uint4*)&ldsA[row * APAD + quad * 8] = vv;
    }
    // stage B tile from WT[n][k]: [128 n x 32 k] — pure b128 traffic now
    #pragma unroll
    for (int it = 0; it < 2; ++it) {
      int slot = it * 256 + tid;
      int row  = slot >> 2;              // n-local
      int quad = slot & 3;
      uint4 vv = *(const uint4*)(WT + (size_t)(n0 + row) * IN_DIM + k0 + quad * 8);
      *(uint4*)&ldsBT[row * BPAD + quad * 8] = vv;
    }
    __syncthreads();

    // B fragments: lane holds col = base + lane%16, K chunks {0..7,16..23}/{8..15,24..31}
    Frag32B bfrag[2];
    #pragma unroll
    for (int nt = 0; nt < 2; ++nt) {
      const unsigned short* p =
          &ldsBT[(wn * 32 + nt * 16 + l16) * BPAD + lhalf * 8];
      bfrag[nt].u[0] = ((const uint4*)p)[0];
      bfrag[nt].u[1] = ((const uint4*)p)[2];
    }
    #pragma unroll
    for (int mt = 0; mt < 4; ++mt) {
      // A fragment: lane row = base + lane%16, K chunks {0..7,16..23}/{8..15,24..31}
      Frag32B afrag;
      const unsigned short* p =
          &ldsA[(wm * 64 + mt * 16 + l16) * APAD + lhalf * 8];
      afrag.u[0] = *(const uint4*)p;
      afrag.u[1] = *(const uint4*)(p + 16);
      #pragma unroll
      for (int nt = 0; nt < 2; ++nt)
        acc[mt][nt] = __builtin_amdgcn_wmma_f32_16x16x32_bf16(
            false, afrag.v, false, bfrag[nt].v, (short)0, acc[mt][nt],
            false, false);
    }
    __syncthreads();
  }

  #pragma unroll
  for (int mt = 0; mt < 4; ++mt) {
    #pragma unroll
    for (int nt = 0; nt < 2; ++nt) {
      int col   = n0 + wn * 32 + nt * 16 + l16;
      float bv  = bias[col];
      int rbase = m0 + wm * 64 + mt * 16 + lhalf * 8;
      #pragma unroll
      for (int v = 0; v < 8; ++v)
        C[(size_t)(rbase + v) * HID_DIM + col] = acc[mt][nt][v] + bv;
    }
  }
}

// ---------------------------------------------------------------------------
// Phase 2: persistent recurrence kernel (64 WGs x 128 thr = 4 wave32).
// WG owns 16 output columns. Recurrent weight slice [3][16][1024] bf16 is
// DMA'd into LDS by the Tensor Data Mover with hardware pad insertion
// (pad 4 DWORDs every 256 DWORDs -> row pitch 1040 bf16, 16B aligned).
// ---------------------------------------------------------------------------
#define NWG  64
#define WCOL 16
#define KROW 1040   // LDS row pitch in bf16 elems (1024 + 2 pads of 8)

__device__ __forceinline__ void grid_barrier(unsigned int* bar, int nwg) {
  __syncthreads();
  if (threadIdx.x == 0) {
    unsigned int gen  = __hip_atomic_load(&bar[1], __ATOMIC_RELAXED,
                                          __HIP_MEMORY_SCOPE_AGENT);
    unsigned int prev = __hip_atomic_fetch_add(&bar[0], 1u, __ATOMIC_ACQ_REL,
                                               __HIP_MEMORY_SCOPE_AGENT);
    if (prev == (unsigned int)(nwg - 1)) {
      __hip_atomic_store(&bar[0], 0u, __ATOMIC_RELAXED,
                         __HIP_MEMORY_SCOPE_AGENT);
      __hip_atomic_store(&bar[1], gen + 1u, __ATOMIC_RELEASE,
                         __HIP_MEMORY_SCOPE_AGENT);
    } else {
      while (__hip_atomic_load(&bar[1], __ATOMIC_ACQUIRE,
                               __HIP_MEMORY_SCOPE_AGENT) == gen)
        __builtin_amdgcn_s_sleep(8);
    }
  }
  __syncthreads();
}

__global__ __launch_bounds__(128) void gru_recurrence(
    const unsigned short* __restrict__ WhT,   // [3][1024 n][1024 k] bf16
    const float* __restrict__ bhr_g, const float* __restrict__ bhz_g,
    const float* __restrict__ bhn_g,
    const float* __restrict__ GR, const float* __restrict__ GZ,
    const float* __restrict__ GN,
    float* __restrict__ h32, unsigned short* __restrict__ hbf,
    float* __restrict__ out, unsigned int* __restrict__ bar)
{
  extern __shared__ unsigned short ldsW[];   // [3*WCOL][KROW]

  const int wg    = blockIdx.x;
  const int tid   = threadIdx.x;
  const int lane  = tid & 31;
  const int wave  = tid >> 5;     // m-tile 0..3 (16 batch rows each)
  const int l16   = lane & 15;
  const int lhalf = lane >> 4;
  const int col0  = wg * WCOL;

  // ---- stage recurrent weight slice into LDS ----
#ifdef USE_TDM
  if (tid < 32) {
    // Tensor DMA descriptor (ISA 8.3-8.6): 3D tile x=1024(k), y=16(col), z=3(gate)
    unsigned long long gaddr =
        (unsigned long long)(size_t)(WhT + (size_t)col0 * HID_DIM);
    unsigned lds_base = (unsigned)(size_t)(void*)ldsW;

    v4ui g0;
    g0[0] = 1u;                                     // count=1, no gather
    g0[1] = lds_base;                               // lds_addr (bytes)
    g0[2] = (unsigned)(gaddr & 0xFFFFFFFFu);        // global_addr[31:0]
    g0[3] = (unsigned)((gaddr >> 32) & 0x01FFFFFFu) // global_addr[56:32]
            | (2u << 30);                           // type = 2 (image)

    v8si g1 = {};
    // data_size=1(2B)@16 | pad_enable@20 | pad_interval=7(256DW)@22
    // | pad_amount=3(4DW=16B)@25
    g1[0] = (1 << 16) | (1 << 20) | (7 << 22) | (3 << 25);
    g1[1] = (int)(1024u << 16);          // tensor_dim0 = 1024 (bits 79:48 lo)
    g1[2] = (int)(1024u << 16);          // dim0 hi=0 | tensor_dim1 = 1024 lo
    g1[3] = (int)(1024u << 16);          // dim1 hi=0 | tile_dim0 = 1024
    g1[4] = (int)(16u | (3u << 16));     // tile_dim1 = 16 | tile_dim2 = 3
    g1[5] = 1024;                        // tensor_dim0_stride lo (= 1024)
    g1[6] = 0;                           // dim0_stride hi | dim1_stride lo
    g1[7] = 16;                          // tensor_dim1_stride hi (1<<20 >> 16)

    v4si g2 = {};
    g2[0] = 3;                           // tensor_dim2 = 3
    v4si g3 = {};

#if __clang_major__ >= 23
    v8si g4 = {};
    __builtin_amdgcn_tensor_load_to_lds(g0, g1, g2, g3, g4, 0);
#else
    __builtin_amdgcn_tensor_load_to_lds(g0, g1, g2, g3, 0);
#endif
    __builtin_amdgcn_s_wait_tensorcnt(0);
  }
#else
  // coalesced manual fallback (weights are pre-transposed: k contiguous)
  for (int e = tid; e < 3 * WCOL * HID_DIM; e += 128) {
    int r = e >> 10;                     // gate*16 + col  (0..47)
    int k = e & 1023;
    ldsW[r * KROW + k + ((k >= 512) ? 8 : 0)] =
        WhT[(size_t)(r >> 4) * HID_DIM * HID_DIM +
            (size_t)(col0 + (r & 15)) * HID_DIM + k];
  }
#endif

  // ---- init h = 0 over our columns (fp32 + both bf16 buffers) ----
  for (int e = tid; e < BATCH * WCOL; e += 128) {
    int b = e >> 4;
    int c = e & (WCOL - 1);
    h32[b * HID_DIM + col0 + c] = 0.0f;
    hbf[b * HID_DIM + col0 + c] = 0;
    hbf[BATCH * HID_DIM + b * HID_DIM + col0 + c] = 0;
  }
  grid_barrier(bar, NWG);

  const int   mycol = col0 + l16;
  const float bhr   = bhr_g[mycol];
  const float bhz   = bhz_g[mycol];
  const float bhn   = bhn_g[mycol];

  const int rowA  = wave * 16 + l16;
  const int kaoff = lhalf * 8;
  const int kboff = lhalf * 8;

  for (int t = 0; t < TSTEPS; ++t) {
    const unsigned short* hcur = hbf + (size_t)(t & 1) * (BATCH * HID_DIM);
    unsigned short*       hnxt = hbf + (size_t)((t + 1) & 1) * (BATCH * HID_DIM);

    v8f acc[3];
    {
      v8f vz = {};
      acc[0] = vz; acc[1] = vz; acc[2] = vz;
    }

    const unsigned short* arow = hcur + (size_t)rowA * HID_DIM;
    #pragma unroll 4
    for (int kc = 0; kc < HID_DIM; kc += 32) {
      Frag32B a;
      a.u[0] = *(const uint4*)(arow + kc + kaoff);
      a.u[1] = *(const uint4*)(arow + kc + kaoff + 16);
      if (kc + 32 < HID_DIM)
        __builtin_prefetch(arow + kc + 32, 0, 0);
      const int kpad = (kc >= 512) ? 8 : 0;   // TDM hardware pad offset
      #pragma unroll
      for (int g = 0; g < 3; ++g) {
        Frag32B b;
        const unsigned short* p =
            &ldsW[(g * WCOL + l16) * KROW + kc + kboff + kpad];
        b.u[0] = ((const uint4*)p)[0];
        b.u[1] = ((const uint4*)p)[2];
        acc[g] = __builtin_amdgcn_wmma_f32_16x16x32_bf16(
            false, a.v, false, b.v, (short)0, acc[g], false, false);
      }
    }

    // fused gating + state update + output store
    #pragma unroll
    for (int v = 0; v < 8; ++v) {
      int b = wave * 16 + lhalf * 8 + v;            // batch row (C layout)
      size_t gidx = ((size_t)b * TSTEPS + t) * HID_DIM + mycol;
      float r = 1.0f / (1.0f + __expf(-(GR[gidx] + acc[0][v] + bhr)));
      float z = 1.0f / (1.0f + __expf(-(GZ[gidx] + acc[1][v] + bhz)));
      float n = tanhf(GN[gidx] + r * (acc[2][v] + bhn));
      float hold = h32[b * HID_DIM + mycol];
      float hnew = (1.0f - z) * n + z * hold;
      h32[b * HID_DIM + mycol] = hnew;
      hnxt[b * HID_DIM + mycol] = f32_to_bf16_raw(hnew);
      out[((size_t)t * BATCH + b) * HID_DIM + mycol] = hnew;
      if (t == TSTEPS - 1)
        out[(size_t)TSTEPS * BATCH * HID_DIM + (size_t)b * HID_DIM + mycol] = hnew;
    }
    grid_barrier(bar, NWG);
  }
}

// ---------------------------------------------------------------------------
// Host-side orchestration
// ---------------------------------------------------------------------------
extern "C" void kernel_launch(void* const* d_in, const int* in_sizes, int n_in,
                              void* d_out, int out_size, void* d_ws,
                              size_t ws_size, hipStream_t stream) {
  (void)in_sizes; (void)n_in; (void)out_size; (void)ws_size;

  const float* inputs = (const float*)d_in[0];
  const float* W_ir = (const float*)d_in[1];  const float* b_ir = (const float*)d_in[2];
  const float* W_hr = (const float*)d_in[3];  const float* b_hr = (const float*)d_in[4];
  const float* W_iz = (const float*)d_in[5];  const float* b_iz = (const float*)d_in[6];
  const float* W_hz = (const float*)d_in[7];  const float* b_hz = (const float*)d_in[8];
  const float* W_in = (const float*)d_in[9];  const float* b_in = (const float*)d_in[10];
  const float* W_hn = (const float*)d_in[11]; const float* b_hn = (const float*)d_in[12];

  char* ws = (char*)d_ws;
  size_t off = 0;
  auto alloc = [&](size_t bytes) -> char* {
    char* p = ws + off;
    off += (bytes + 255) & ~(size_t)255;
    return p;
  };

  unsigned short* x_bf    = (unsigned short*)alloc((size_t)MROWS * IN_DIM * 2);
  unsigned short* WirT_bf = (unsigned short*)alloc((size_t)IN_DIM * HID_DIM * 2);
  unsigned short* WizT_bf = (unsigned short*)alloc((size_t)IN_DIM * HID_DIM * 2);
  unsigned short* WinT_bf = (unsigned short*)alloc((size_t)IN_DIM * HID_DIM * 2);
  unsigned short* WhT_bf  = (unsigned short*)alloc((size_t)3 * HID_DIM * HID_DIM * 2);
  float* GR  = (float*)alloc((size_t)MROWS * HID_DIM * 4);
  float* GZ  = (float*)alloc((size_t)MROWS * HID_DIM * 4);
  float* GN  = (float*)alloc((size_t)MROWS * HID_DIM * 4);
  float* h32 = (float*)alloc((size_t)BATCH * HID_DIM * 4);
  unsigned short* hbf = (unsigned short*)alloc((size_t)2 * BATCH * HID_DIM * 2);
  unsigned int* bar = (unsigned int*)alloc(256);

  hipMemsetAsync(bar, 0, 256, stream);

  cvt_f32_bf16<<<2048, 256, 0, stream>>>(inputs, x_bf, MROWS * IN_DIM);

  dim3 tgrid(HID_DIM / 32, IN_DIM / 32);
  dim3 tblk(32, 8);
  transp_cvt_f32_bf16<<<tgrid, tblk, 0, stream>>>(W_ir, WirT_bf);
  transp_cvt_f32_bf16<<<tgrid, tblk, 0, stream>>>(W_iz, WizT_bf);
  transp_cvt_f32_bf16<<<tgrid, tblk, 0, stream>>>(W_in, WinT_bf);
  transp_cvt_f32_bf16<<<tgrid, tblk, 0, stream>>>(
      W_hr, WhT_bf + (size_t)0 * HID_DIM * HID_DIM);
  transp_cvt_f32_bf16<<<tgrid, tblk, 0, stream>>>(
      W_hz, WhT_bf + (size_t)1 * HID_DIM * HID_DIM);
  transp_cvt_f32_bf16<<<tgrid, tblk, 0, stream>>>(
      W_hn, WhT_bf + (size_t)2 * HID_DIM * HID_DIM);

  dim3 g1(HID_DIM / BN, MROWS / BM, 3);
  gemm_xproj<<<g1, 256, 0, stream>>>(x_bf, WirT_bf, WizT_bf, WinT_bf,
                                     b_ir, b_iz, b_in, GR, GZ, GN);

  size_t lds_bytes = (size_t)3 * WCOL * KROW * sizeof(unsigned short); // 99840
  gru_recurrence<<<NWG, 128, lds_bytes, stream>>>(
      WhT_bf, b_hr, b_hz, b_hn, GR, GZ, GN,
      h32, hbf, (float*)d_out, bar);
}
